// WindowAttention_14104672600652
// MI455X (gfx1250) — matrix operand order repaired and checked by reading, not compiled, not run on hardware
//
#include <hip/hip_runtime.h>

typedef __attribute__((ext_vector_type(16))) _Float16 v16h;
typedef __attribute__((ext_vector_type(8)))  _Float16 v8h;
typedef __attribute__((ext_vector_type(8)))  float    v8f;

#define NTOK 49      // tokens per window
#define MPAD 64      // padded tokens
#define DIMC 192     // channels
#define NH   6       // heads
#define HD   32      // head dim
#define NWIN 64      // mask windows

// LDS row strides (in halves / floats), all 16-byte aligned
#define XS 200       // xh / outh row stride (halves)  64*200*2 = 25600 B
#define QS 400       // qk  row stride (halves)        64*400*2 = 51200 B
#define VS 72        // vt  row stride (halves)       192*72*2  = 27648 B
#define AS 68        // attn f32 row stride (floats)   64*68*4  = 17408 B
#define BS 72        // attn f16 row stride (halves)   64*72*2  =  9216 B

#define OFF_XH   0
#define OFF_QK   25600
#define OFF_VT   76800
#define OFF_AF   104448
#define OFF_A16  121856
#define SMEM_BYTES 131072

static __device__ inline v16h ldpair(const _Float16* p0, const _Float16* p1) {
    v8h a = *(const v8h*)p0;
    v8h b = *(const v8h*)p1;
    v16h r;
#pragma unroll
    for (int j = 0; j < 8; ++j) { r[j] = a[j]; r[j + 8] = b[j]; }
    return r;
}

static __device__ inline v8f wmma16(v16h a, v16h b, v8f c) {
    return __builtin_amdgcn_wmma_f32_16x16x32_f16(false, a, false, b, (short)0, c,
                                                  false, false);
}

__global__ void convert_weights_kernel(const float* __restrict__ qkv_w,
                                       const float* __restrict__ proj_w,
                                       _Float16* __restrict__ Wh,
                                       _Float16* __restrict__ Ph) {
    int i = blockIdx.x * 256 + threadIdx.x;
    const int NQ = 3 * DIMC * DIMC;   // 110592
    const int NP = DIMC * DIMC;       // 36864
    if (i < NQ) Wh[i] = (_Float16)qkv_w[i];
    else if (i < NQ + NP) Ph[i - NQ] = (_Float16)proj_w[i - NQ];
}

__global__ __launch_bounds__(256) void swin_window_attn_kernel(
    const float* __restrict__ x,          // [4096,49,192]
    const float* __restrict__ mask,       // [64,49,49]
    const float* __restrict__ qkv_b,      // [576]
    const float* __restrict__ proj_b,     // [192]
    const float* __restrict__ bias_table, // [169,6]
    const int*   __restrict__ rel_index,  // [49,49]
    const _Float16* __restrict__ Wh,      // [576,192] f16 (row-major = B frag layout)
    const _Float16* __restrict__ Ph,      // [192,192] f16
    float* __restrict__ out)              // [4096,49,192]
{
    extern __shared__ char smem[];
    _Float16* xh   = (_Float16*)(smem + OFF_XH);   // 64 x XS   (x, later reused as outh)
    _Float16* qkh  = (_Float16*)(smem + OFF_QK);   // 64 x QS   cols 0..191 q*scale, 192..383 k
    _Float16* vt   = (_Float16*)(smem + OFF_VT);   // 192 x VS  v transposed [channel][token]
    float*    attf = (float*)   (smem + OFF_AF);   // 64 x AS   raw scores
    _Float16* at16 = (_Float16*)(smem + OFF_A16);  // 64 x BS   softmax probs (f16)
    _Float16* outh = xh;

    const int tid  = threadIdx.x;
    const int lane = tid & 31;
    const int wave = tid >> 5;
    const int ln   = lane & 15;   // fragment column index
    const int hi   = lane >> 4;   // fragment half select
    const int win  = blockIdx.x;

    // ---- Stage A: x -> LDS f16 (pad rows 49..63 with zero) ----
    {
        const float* xb = x + (size_t)win * (NTOK * DIMC);
        for (int i = tid; i < MPAD * DIMC; i += 256) {
            int r = i / DIMC, c = i - r * DIMC;
            float v = (r < NTOK) ? xb[r * DIMC + c] : 0.f;
            xh[r * XS + c] = (_Float16)v;
        }
    }
    __syncthreads();

    // ---- Stage B: QKV gemm  [64x192] @ [192x576] + bias ----
    for (int t = wave; t < 144; t += 8) {
        int m0 = (t & 3) * 16;
        int n0 = (t >> 2) * 16;
        float binit = qkv_b[n0 + ln];
        v8f acc;
#pragma unroll
        for (int r = 0; r < 8; ++r) acc[r] = binit;
#pragma unroll
        for (int ks = 0; ks < 6; ++ks) {
            int k0 = ks * 32;
            const _Float16* ar = xh + (m0 + ln) * XS + k0 + 8 * hi;
            v16h a = ldpair(ar, ar + 16);
            const _Float16* br = Wh + (size_t)(n0 + ln) * DIMC + k0 + 8 * hi;
            v16h b = ldpair(br, br + 16);
            acc = wmma16(a, b, acc);
        }
        if (n0 < 2 * DIMC) {           // q (scaled) or k -> qkh
            float sc = (n0 < DIMC) ? 0.1767766953f : 1.0f;
#pragma unroll
            for (int r = 0; r < 8; ++r)
                qkh[(m0 + r + 8 * hi) * QS + n0 + ln] = (_Float16)(acc[r] * sc);
        } else {                       // v -> vt transposed [channel][token]
#pragma unroll
            for (int r = 0; r < 8; ++r)
                vt[(n0 - 2 * DIMC + ln) * VS + (m0 + r + 8 * hi)] = (_Float16)acc[r];
        }
    }
    __syncthreads();

    // ---- Per-head attention ----
    const int wgrp = win & (NWIN - 1);
    for (int h = 0; h < NH; ++h) {
        // scores: q @ k^T + rel_bias + mask   (16 tiles, 2 per wave)
        for (int t = wave; t < 16; t += 8) {
            int m0 = (t & 3) * 16;
            int n0 = (t >> 2) * 16;
            v8f acc;
#pragma unroll
            for (int r = 0; r < 8; ++r) {
                int m = m0 + r + 8 * hi;
                int n = n0 + ln;
                int mm = (m < NTOK) ? m : (NTOK - 1);
                int nn = (n < NTOK) ? n : (NTOK - 1);
                int ridx = rel_index[mm * NTOK + nn];
                acc[r] = bias_table[ridx * NH + h] +
                         mask[((size_t)wgrp * NTOK + mm) * NTOK + nn];
            }
            const _Float16* ar = qkh + (m0 + ln) * QS + h * HD + 8 * hi;
            v16h a = ldpair(ar, ar + 16);
            const _Float16* br = qkh + (n0 + ln) * QS + DIMC + h * HD + 8 * hi;
            v16h b = ldpair(br, br + 16);
            acc = wmma16(a, b, acc);
#pragma unroll
            for (int r = 0; r < 8; ++r)
                attf[(m0 + r + 8 * hi) * AS + n0 + ln] = acc[r];
        }
        __syncthreads();

        // softmax over valid 49 columns; zero padded columns
        if (tid < MPAD) {
            const float* row = attf + tid * AS;
            _Float16* prow = at16 + tid * BS;
            float mx = row[0];
            for (int j = 1; j < NTOK; ++j) mx = fmaxf(mx, row[j]);
            float s = 0.f;
            for (int j = 0; j < NTOK; ++j) {
                float ev = __expf(row[j] - mx);
                s += ev;
                prow[j] = (_Float16)ev;
            }
            float inv = 1.f / s;
            for (int j = 0; j < NTOK; ++j)
                prow[j] = (_Float16)((float)prow[j] * inv);
            for (int j = NTOK; j < MPAD; ++j) prow[j] = (_Float16)0.f;
        }
        __syncthreads();

        // attn @ v  (8 tiles, 1 per wave, K=64 -> 2 wmma)
        {
            int m0 = (wave & 3) * 16;
            int n0 = (wave >> 2) * 16;
            v8f acc;
#pragma unroll
            for (int r = 0; r < 8; ++r) acc[r] = 0.f;
#pragma unroll
            for (int ks = 0; ks < 2; ++ks) {
                int k0 = ks * 32;
                const _Float16* ar = at16 + (m0 + ln) * BS + k0 + 8 * hi;
                v16h a = ldpair(ar, ar + 16);
                const _Float16* br = vt + (h * HD + n0 + ln) * VS + k0 + 8 * hi;
                v16h b = ldpair(br, br + 16);
                acc = wmma16(a, b, acc);
            }
#pragma unroll
            for (int r = 0; r < 8; ++r)
                outh[(m0 + r + 8 * hi) * XS + h * HD + n0 + ln] = (_Float16)acc[r];
        }
        __syncthreads();
    }

    // ---- Proj: [64x192] @ [192x192] + bias, store tokens < 49 ----
    {
        float* ob = out + (size_t)win * (NTOK * DIMC);
        for (int t = wave; t < 48; t += 8) {
            int m0 = (t & 3) * 16;
            int n0 = (t >> 2) * 16;
            float binit = proj_b[n0 + ln];
            v8f acc;
#pragma unroll
            for (int r = 0; r < 8; ++r) acc[r] = binit;
#pragma unroll
            for (int ks = 0; ks < 6; ++ks) {
                int k0 = ks * 32;
                const _Float16* ar = outh + (m0 + ln) * XS + k0 + 8 * hi;
                v16h a = ldpair(ar, ar + 16);
                const _Float16* br = Ph + (size_t)(n0 + ln) * DIMC + k0 + 8 * hi;
                v16h b = ldpair(br, br + 16);
                acc = wmma16(a, b, acc);
            }
#pragma unroll
            for (int r = 0; r < 8; ++r) {
                int m = m0 + r + 8 * hi;
                if (m < NTOK) ob[m * DIMC + n0 + ln] = acc[r];
            }
        }
    }
}

extern "C" void kernel_launch(void* const* d_in, const int* in_sizes, int n_in,
                              void* d_out, int out_size, void* d_ws, size_t ws_size,
                              hipStream_t stream) {
    const float* x          = (const float*)d_in[0];
    const float* mask       = (const float*)d_in[1];
    const float* qkv_w      = (const float*)d_in[2];
    const float* qkv_b      = (const float*)d_in[3];
    const float* proj_w     = (const float*)d_in[4];
    const float* proj_b     = (const float*)d_in[5];
    const float* bias_table = (const float*)d_in[6];
    const int*   rel_index  = (const int*)d_in[7];
    float* out = (float*)d_out;

    _Float16* Wh = (_Float16*)d_ws;                 // 576*192 halves = 221184 B
    _Float16* Ph = Wh + 3 * DIMC * DIMC;            // 192*192 halves =  73728 B

    const int conv_elems = 3 * DIMC * DIMC + DIMC * DIMC; // 147456
    convert_weights_kernel<<<(conv_elems + 255) / 256, 256, 0, stream>>>(
        qkv_w, proj_w, Wh, Ph);

    (void)hipFuncSetAttribute((const void*)swin_window_attn_kernel,
                              hipFuncAttributeMaxDynamicSharedMemorySize,
                              SMEM_BYTES);

    swin_window_attn_kernel<<<4096, 256, SMEM_BYTES, stream>>>(
        x, mask, qkv_b, proj_b, bias_table, rel_index, Wh, Ph, out);
}
